// XAIguiAttention_87806311399716
// MI455X (gfx1250) — matrix operand order repaired and checked by reading, not compiled, not run on hardware
//
#include <hip/hip_runtime.h>
#include <stdint.h>

typedef __attribute__((ext_vector_type(16))) __bf16 v16bf;
typedef __attribute__((ext_vector_type(8)))  float  v8f;
typedef unsigned short u16;

__device__ __forceinline__ u16 f32_to_bf16(float f) {
  unsigned int u = __float_as_uint(f);
  unsigned int r = 0x7FFFu + ((u >> 16) & 1u);
  return (u16)((u + r) >> 16);
}
__device__ __forceinline__ float bf16_to_f32(u16 h) {
  return __uint_as_float(((unsigned int)h) << 16);
}

union Frag16 { uint4 u[2]; v16bf v; };

// generic (flat) pointer -> raw LDS byte offset (addrspacecast to AS3, then ptrtoint)
__device__ __forceinline__ unsigned lds_off_of(const void* p) {
  return (unsigned)(unsigned long long)(__attribute__((address_space(3))) const void*)p;
}

// CDNA5 async DMA: LDS[vdst] = MEM[vaddr], 16 bytes per lane, tracked by ASYNCcnt
__device__ __forceinline__ void async_load_b128(unsigned lds_byte_off, const u16* gptr) {
  asm volatile("global_load_async_to_lds_b128 %0, %1, off"
               :: "v"(lds_byte_off), "v"((unsigned long long)(uintptr_t)gptr)
               : "memory");
}
__device__ __forceinline__ void wait_asynccnt0() {
  asm volatile("s_wait_asynccnt 0" ::: "memory");
}

#define BM 128
#define BN 128
#define BK 32
#define LDK 40   // padded LDS row stride (halfwords): 80B -> conflict-free, 16B aligned

// C[M,N] = A[M,K](bf16,row-major) * Bt[N,K](bf16, i.e. B transposed)
// OUT_F32: write f32 + bias ; else write bf16
template<bool OUT_F32>
__global__ __launch_bounds__(256)
void gemm_bf16_wmma(const u16* __restrict__ A, const u16* __restrict__ Bt,
                    const float* __restrict__ bias, void* __restrict__ Cout,
                    int M, int N, int K)
{
  __shared__ __align__(16) u16 As[2][BM * LDK];
  __shared__ __align__(16) u16 Bs[2][BN * LDK];

  const int tid  = threadIdx.x;
  const int lane = tid & 31;
  const int wave = tid >> 5;
  const int wm = (wave >> 2) * 64;   // 2 waves along M
  const int wn = (wave & 3) * 32;    // 4 waves along N
  const int m0 = blockIdx.y * BM;
  const int n0 = blockIdx.x * BN;

  // staging map: 256 threads cover 128 rows x 32 halfwords (16 hw = 2 x b128 per thread)
  const int sr = tid >> 1;
  const int sk = (tid & 1) << 4;
  const int so = sr * LDK + sk;

  const u16* Ap = A  + (size_t)(m0 + sr) * K + sk;
  const u16* Bp = Bt + (size_t)(n0 + sr) * K + sk;

  // async-stage tile 0 into buffer 0
  async_load_b128(lds_off_of(&As[0][so]),     Ap);
  async_load_b128(lds_off_of(&As[0][so + 8]), Ap + 8);
  async_load_b128(lds_off_of(&Bs[0][so]),     Bp);
  async_load_b128(lds_off_of(&Bs[0][so + 8]), Bp + 8);
  wait_asynccnt0();
  __syncthreads();

  v8f acc[4][2] = {};

  const int lhalf = lane >> 4;   // 0/1 -> which K-half this lane holds
  const int lmod  = lane & 15;
  const int KT = K / BK;
  int buf = 0;

  for (int kt = 0; kt < KT; ++kt) {
    if (kt + 1 < KT) {
      // prefetch tile kt+1 into the other buffer; DMA overlaps WMMA below
      const int nb = buf ^ 1;
      const u16* An = Ap + (size_t)(kt + 1) * BK;
      const u16* Bn = Bp + (size_t)(kt + 1) * BK;
      async_load_b128(lds_off_of(&As[nb][so]),     An);
      async_load_b128(lds_off_of(&As[nb][so + 8]), An + 8);
      async_load_b128(lds_off_of(&Bs[nb][so]),     Bn);
      async_load_b128(lds_off_of(&Bs[nb][so + 8]), Bn + 8);
    }

    Frag16 af[4], bfr[2];
    // A frag (16x32 bf16): lanes<16 hold K0..7 & K16..23, lanes>=16 hold K8..15 & K24..31
#pragma unroll
    for (int mi = 0; mi < 4; ++mi) {
      const int base = (wm + mi * 16 + lmod) * LDK;
      af[mi].u[0] = *(const uint4*)&As[buf][base + lhalf * 8];
      af[mi].u[1] = *(const uint4*)&As[buf][base + 16 + lhalf * 8];
    }
    // B frag (32x16): lane = N, lanes<16 hold K0..7 & K16..23, lanes>=16 hold K8..15 & K24..31
#pragma unroll
    for (int ni = 0; ni < 2; ++ni) {
      const int base = (wn + ni * 16 + lmod) * LDK;
      bfr[ni].u[0] = *(const uint4*)&Bs[buf][base + lhalf * 8];
      bfr[ni].u[1] = *(const uint4*)&Bs[buf][base + 16 + lhalf * 8];
    }

#pragma unroll
    for (int mi = 0; mi < 4; ++mi)
#pragma unroll
      for (int ni = 0; ni < 2; ++ni)
        acc[mi][ni] = __builtin_amdgcn_wmma_f32_16x16x32_bf16(
            false, af[mi].v, false, bfr[ni].v, (short)0, acc[mi][ni], false, false);

    if (kt + 1 < KT) {
      wait_asynccnt0();     // this wave's DMA for tile kt+1 done
      __syncthreads();      // everyone's DMA visible
      buf ^= 1;
    }
  }

  // epilogue: C layout -> VGPR j, lane l : row = j + 8*(l>=16), col = l%16
#pragma unroll
  for (int mi = 0; mi < 4; ++mi) {
#pragma unroll
    for (int ni = 0; ni < 2; ++ni) {
      const int c = n0 + wn + ni * 16 + lmod;
      float bv = 0.0f;
      if (OUT_F32) bv = bias[c];
#pragma unroll
      for (int j = 0; j < 8; ++j) {
        const int r = m0 + wm + mi * 16 + lhalf * 8 + j;
        const float v = acc[mi][ni][j];
        if (OUT_F32)
          ((float*)Cout)[(size_t)r * N + c] = v + bv;
        else
          ((u16*)Cout)[(size_t)r * N + c] = f32_to_bf16(v);
      }
    }
  }
}

__constant__ float c_blo[9] = {1.f, 4.f, 8.f, 10.f, 12.f, 30.f, 1.f, 8.f, 1.f};
__constant__ float c_bhi[9] = {4.f, 8.f, 10.f, 12.f, 30.f, 45.f, 8.f, 30.f, 45.f};

// one thread per (b,n,s in {q,k},h, pair). Pair (2i,2i+1) shares one angle.
__global__ __launch_bounds__(256)
void rotary_kernel(u16* __restrict__ QKV, const float* __restrict__ demo)
{
  const int p  = blockIdx.x * 256 + threadIdx.x;
  const int dp = p & 31;               // pair index 0..31 -> d = 2*dp
  int t = p >> 5;
  const int h = t % 12; t /= 12;
  const int s = t & 1;  t >>= 1;
  const int n = t % 9;
  const int b = t / 9;

  // freqs = pi * linspace(1, 5, 16); d<32 uses lo=band[n][0], d>=32 uses hi
  const float fidx  = (dp < 16) ? (float)dp : (float)(dp - 16);
  const float bound = (dp < 16) ? c_blo[n] : c_bhi[n];
  const float ang   = bound * 3.14159265358979f * (1.0f + (4.0f / 15.0f) * fidx);
  float sn, cs;
  __sincosf(ang, &sn, &cs);

  const float age = demo[2 * b];
  const float gen = demo[2 * b + 1];

  u16* ptr = QKV + ((size_t)(b * 9 + n)) * 2304 + s * 768 + h * 64 + dp * 2;
  const unsigned int pk = *(const unsigned int*)ptr;
  const float x1 = bf16_to_f32((u16)(pk & 0xFFFFu));
  const float x2 = bf16_to_f32((u16)(pk >> 16));
  const float y1 = age * x1 * cs - gen * x2 * sn;
  const float y2 = age * x2 * cs + gen * x1 * sn;
  *(unsigned int*)ptr = (unsigned int)f32_to_bf16(y1) |
                        ((unsigned int)f32_to_bf16(y2) << 16);
}

// one wave per (b,h); 8 heads per 256-thread block
__global__ __launch_bounds__(256)
void attention_kernel(const u16* __restrict__ QKV, u16* __restrict__ AO)
{
  __shared__ float sq[8][576];
  __shared__ float sk[8][576];
  __shared__ float sv[8][576];
  __shared__ float sp[8][96];

  const int wave = threadIdx.x >> 5;
  const int lane = threadIdx.x & 31;
  const int gh = blockIdx.x * 8 + wave;
  const int b = gh / 12;
  const int h = gh - b * 12;

  const u16* base = QKV + (size_t)b * 9 * 2304 + h * 64;
#pragma unroll
  for (int idx = lane; idx < 576; idx += 32) {
    const int n = idx >> 6, d = idx & 63;
    const size_t ro = (size_t)n * 2304 + d;
    sq[wave][idx] = bf16_to_f32(base[ro]);
    sk[wave][idx] = bf16_to_f32(base[ro + 768]);
    sv[wave][idx] = bf16_to_f32(base[ro + 1536]);
  }
  __syncthreads();

  for (int idx = lane; idx < 81; idx += 32) {
    const int i = idx / 9, j = idx - i * 9;
    float s = 0.f;
#pragma unroll
    for (int d = 0; d < 64; ++d)
      s += sq[wave][i * 64 + d] * sk[wave][j * 64 + d];
    sp[wave][i * 9 + j] = s * 0.125f;   // 64^-0.5
  }
  __syncthreads();

  if (lane < 9) {
    float mx = -3.0e38f;
    for (int j = 0; j < 9; ++j) mx = fmaxf(mx, sp[wave][lane * 9 + j]);
    float e[9], sum = 0.f;
    for (int j = 0; j < 9; ++j) { e[j] = __expf(sp[wave][lane * 9 + j] - mx); sum += e[j]; }
    const float inv = 1.0f / sum;
    for (int j = 0; j < 9; ++j) sp[wave][lane * 9 + j] = e[j] * inv;
  }
  __syncthreads();

  u16* aout = AO + (size_t)b * 9 * 768 + h * 64;
  for (int idx = lane; idx < 576; idx += 32) {
    const int i = idx >> 6, d = idx & 63;
    float o = 0.f;
#pragma unroll
    for (int j = 0; j < 9; ++j)
      o += sp[wave][i * 9 + j] * sv[wave][j * 64 + d];
    aout[(size_t)i * 768 + d] = f32_to_bf16(o);
  }
}

__global__ __launch_bounds__(256)
void cvt_f32_bf16(const float* __restrict__ src, u16* __restrict__ dst, int n)
{
  const int i = blockIdx.x * 256 + threadIdx.x;
  if (i < n) dst[i] = f32_to_bf16(src[i]);
}

// src [K,N] f32 -> dst [N,K] bf16 (transposed)
__global__ __launch_bounds__(256)
void cvt_transpose_bf16(const float* __restrict__ src, u16* __restrict__ dst, int K, int N)
{
  const int i = blockIdx.x * 256 + threadIdx.x;
  if (i < K * N) {
    const int k = i / N, n = i - k * N;
    dst[(size_t)n * K + k] = f32_to_bf16(src[i]);
  }
}

extern "C" void kernel_launch(void* const* d_in, const int* in_sizes, int n_in,
                              void* d_out, int out_size, void* d_ws, size_t ws_size,
                              hipStream_t stream) {
  (void)in_sizes; (void)n_in; (void)out_size; (void)ws_size;
  const float* x     = (const float*)d_in[0];
  const float* demo  = (const float*)d_in[1];
  const float* Wqkv  = (const float*)d_in[2];
  const float* Wproj = (const float*)d_in[3];
  const float* bproj = (const float*)d_in[4];
  float* out = (float*)d_out;

  const int Bb = 4096, Nn = 9, C = 768, H = 12;
  const int M  = Bb * Nn;   // 36864
  const int K  = C;         // 768
  const int N1 = 3 * C;     // 2304

  u16* Xb   = (u16*)d_ws;                 // M*K
  u16* Wqt  = Xb   + (size_t)M * K;       // N1*K (transposed)
  u16* Wpt  = Wqt  + (size_t)N1 * K;      // C*K  (transposed)
  u16* QKVb = Wpt  + (size_t)C * K;       // M*N1
  u16* AOb  = QKVb + (size_t)M * N1;      // M*C

  cvt_f32_bf16<<<(M * K + 255) / 256, 256, 0, stream>>>(x, Xb, M * K);
  cvt_transpose_bf16<<<(K * N1 + 255) / 256, 256, 0, stream>>>(Wqkv, Wqt, K, N1);
  cvt_transpose_bf16<<<(K * C + 255) / 256, 256, 0, stream>>>(Wproj, Wpt, K, C);

  gemm_bf16_wmma<false><<<dim3(N1 / BN, M / BM), 256, 0, stream>>>(
      Xb, Wqt, nullptr, (void*)QKVb, M, N1, K);

  const int pairs = Bb * Nn * 2 * H * 32;   // 28,311,552
  rotary_kernel<<<pairs / 256, 256, 0, stream>>>(QKVb, demo);

  attention_kernel<<<(Bb * H) / 8, 256, 0, stream>>>(QKVb, AOb);

  gemm_bf16_wmma<true><<<dim3(C / BN, M / BM), 256, 0, stream>>>(
      AOb, Wpt, bproj, (void*)out, M, C, K);
}
